// MRNNFixD_62354335203967
// MI455X (gfx1250) — matrix operand order repaired and checked by reading, not compiled, not run on hardware
//
#include <hip/hip_runtime.h>
#include <hip/hip_bf16.h>
#include <math.h>

#define T_LEN 512
#define BATCH 128
#define HID   256
#define G4H   (4*HID)
#define KMEM  25
#define NOUT  5

typedef __attribute__((ext_vector_type(16))) _Float16 v16h;
typedef __attribute__((ext_vector_type(8)))  _Float16 v8h;
typedef __attribute__((ext_vector_type(8)))  float    v8f;

// ---- global workspace layout (bytes, all 256-aligned) ----
#define WS_WD    0u                       // [KMEM,HID] f32 binomial weights
#define WS_WXB   25600u                   // [HID,HID]  f16 w_x^T  (fragment-packed)
#define WS_UB    (WS_WXB  + 131072u)      // [HID,HID]  f16 u_h^T  (fragment-packed)
#define WS_IHB   (WS_UB   + 131072u)      // [HID,4H]   f16 w_ih^T (fragment-packed)
#define WS_HHB   (WS_IHB  + 524288u)      // [HID,4H]   f16 w_hh^T (fragment-packed)
#define WS_RING  (WS_HHB  + 524288u)      // [KMEM,B,H] f16 hidden ring
#define WS_CX    (WS_RING + 1638400u)     // [B,H] f32
#define WS_GATES (WS_CX   + 131072u)      // [B,4H] f32
#define WS_SEL   (WS_GATES+ 524288u)      // [B,H] f32

// ---- LDS layout for the persistent scan (192 KB of the WGP's 320 KB) ----
#define SM_MEM 0u            // [B,H] f16
#define SM_H1  65536u        // [B,H] f16
#define SM_HX  131072u       // [B,H] f16
#define SM_TOTAL 196608u

__device__ __forceinline__ float sigm(float x) { return 1.0f / (1.0f + __expf(-x)); }

__device__ __forceinline__ v8f wmma_f16(v16h a, v16h b, v8f c) {
  return __builtin_amdgcn_wmma_f32_16x16x32_f16(false, a, false, b, (short)0, c,
                                                false, false);
}

// A-matrix (16x32 f16) fragment gather, ISA 7.12.2 wave32 layout.
// Elements 0-7 and 8-15 are each 16B-contiguous -> two b128 loads.
__device__ __forceinline__ v16h load_a_f16(const _Float16* A, int lda, int row0,
                                           int kbase, int lane) {
  const int half = lane >> 4, m = lane & 15;
  const _Float16* rp = A + (row0 + m) * lda + kbase + 8 * half;
  v16h a;
#pragma unroll
  for (int e = 0; e < 16; ++e) a[e] = rp[(e & 7) + ((e >> 3) << 4)];
  return a;
}

__device__ __forceinline__ v16h load_a_f32(const float* A, int lda, int row0,
                                           int kbase, int lane) {
  const int half = lane >> 4, m = lane & 15;
  const float* rp = A + (size_t)(row0 + m) * lda + kbase + 8 * half;
  v16h a;
#pragma unroll
  for (int e = 0; e < 16; ++e) a[e] = (_Float16)rp[(e & 7) + ((e >> 3) << 4)];
  return a;
}

// B fragment from fragment-packed weights: each 32x16 K-panel is a 1 KB block,
// lane's 16 f16 contiguous -> two global_load_b128, fully coalesced.
// block = (n0/16)*(HID/32) + (kbase/32)
__device__ __forceinline__ v16h load_b_packed(const _Float16* B, int n0, int kbase,
                                              int lane) {
  const _Float16* p =
      B + ((size_t)((n0 >> 4) * (HID / 32) + (kbase >> 5)) << 9) + (lane << 4);
  v16h b;
#pragma unroll
  for (int e = 0; e < 16; ++e) b[e] = p[e];
  return b;
}

// ------------------------------------------------------------------
// Kernel 1: per-call prep — wd weights, fragment-packed f16 weights, zero state
// ------------------------------------------------------------------
__global__ void mrnn_prep(const float* __restrict__ bd1, const float* __restrict__ w_x,
                          const float* __restrict__ u_h, const float* __restrict__ w_ih,
                          const float* __restrict__ w_hh, char* __restrict__ ws) {
  const int gid = blockIdx.x * blockDim.x + threadIdx.x;
  const int gsz = gridDim.x * blockDim.x;
  float*    wd   = (float*)(ws + WS_WD);
  _Float16* wxB  = (_Float16*)(ws + WS_WXB);
  _Float16* uB   = (_Float16*)(ws + WS_UB);
  _Float16* ihB  = (_Float16*)(ws + WS_IHB);
  _Float16* hhB  = (_Float16*)(ws + WS_HHB);
  _Float16* ring = (_Float16*)(ws + WS_RING);
  float*    cx   = (float*)(ws + WS_CX);

  // fractional-difference weights: wd[K-1-i] = ws_{i+1}, ws_{i+1}=ws_i*(i-d)/(i+1)
  for (int h = gid; h < HID; h += gsz) {
    float d = 0.5f * sigm(bd1[h]);
    float w = 1.0f;
    for (int i = 0; i < KMEM; ++i) {
      w = w * ((float)i - d) / ((float)i + 1.0f);
      wd[(KMEM - 1 - i) * HID + h] = w;
    }
  }
  // fragment-pack: dst[block*512 + lane*16 + e] = W[(nt*16+n)*256 + kb*32 + 8*half + (e&7) + 16*(e>>3)]
  for (int idx = gid; idx < HID * HID; idx += gsz) {
    int block = idx >> 9, r = idx & 511;
    int lane = r >> 4, e = r & 15;
    int nt = block >> 3, kb = block & 7;
    int half = lane >> 4, n = lane & 15;
    int src = (nt * 16 + n) * HID + kb * 32 + 8 * half + (e & 7) + ((e >> 3) << 4);
    wxB[idx] = (_Float16)w_x[src];
    uB[idx]  = (_Float16)u_h[src];
  }
  for (int idx = gid; idx < HID * G4H; idx += gsz) {
    int block = idx >> 9, r = idx & 511;
    int lane = r >> 4, e = r & 15;
    int nt = block >> 3, kb = block & 7;
    int half = lane >> 4, n = lane & 15;
    int src = (nt * 16 + n) * HID + kb * 32 + 8 * half + (e & 7) + ((e >> 3) << 4);
    ihB[idx] = (_Float16)w_ih[src];
    hhB[idx] = (_Float16)w_hh[src];
  }
  // zero recurrent state (fresh every call -> deterministic graph replay)
  for (int idx = gid; idx < KMEM * BATCH * HID; idx += gsz) ring[idx] = (_Float16)0.0f;
  for (int idx = gid; idx < BATCH * HID; idx += gsz) cx[idx] = 0.0f;
}

// ------------------------------------------------------------------
// Kernel 2: persistent sequential scan, one workgroup = 32 wave32 waves on 1 WGP.
// mem / h1 / hx live in LDS; weights (f16, fragment-packed) stay L2-resident.
// ------------------------------------------------------------------
__global__ void __launch_bounds__(1024, 1)
mrnn_scan(const float* __restrict__ inputs, const int* __restrict__ length,
          const float* __restrict__ b_m, const float* __restrict__ b_ih,
          const float* __restrict__ b_hh, char* __restrict__ ws) {
  extern __shared__ char smem[];
  _Float16* memb = (_Float16*)(smem + SM_MEM);
  _Float16* h1   = (_Float16*)(smem + SM_H1);
  _Float16* hx   = (_Float16*)(smem + SM_HX);

  const int tid  = threadIdx.x;
  const int wave = tid >> 5;
  const int lane = tid & 31;

  const float*    wd    = (const float*)(ws + WS_WD);
  const _Float16* wxB   = (const _Float16*)(ws + WS_WXB);
  const _Float16* uB    = (const _Float16*)(ws + WS_UB);
  const _Float16* ihB   = (const _Float16*)(ws + WS_IHB);
  const _Float16* hhB   = (const _Float16*)(ws + WS_HHB);
  _Float16*       ring  = (_Float16*)(ws + WS_RING);
  float*          cx    = (float*)(ws + WS_CX);
  float*          gates = (float*)(ws + WS_GATES);
  float*          sel   = (float*)(ws + WS_SEL);

  // zero hx in LDS (prev hidden state of the LSTM)
  for (int g = tid; g < (BATCH * HID) / 8; g += 1024)
    *(v8h*)(hx + g * 8) = (v8h)(_Float16)0.0f;
  __syncthreads();

  for (int t = 0; t < T_LEN; ++t) {
    // ---- Phase 0: mem[b,h] = sum_k wd[k,h]*hid[k,b,h]; ring slot t+k, minus K once if >= K.
    // 8-wide vectorized: each lane owns 8 consecutive h -> coalesced b128s.
    for (int c0 = 0; c0 < (BATCH * HID) / (1024 * 8); ++c0) {
      int base = (tid + c0 * 1024) * 8;
      int h0 = base & (HID - 1);
      float acc[8] = {0.f, 0.f, 0.f, 0.f, 0.f, 0.f, 0.f, 0.f};
#pragma unroll
      for (int k = 0; k < KMEM; ++k) {
        int slot = t + k;
        slot -= (slot >= KMEM) ? KMEM : 0;
        v8h rv = *(const v8h*)(ring + slot * (BATCH * HID) + base);
        const float* wp = wd + k * HID + h0;
#pragma unroll
        for (int e = 0; e < 8; ++e) acc[e] += wp[e] * (float)rv[e];
      }
      v8h out;
#pragma unroll
      for (int e = 0; e < 8; ++e) out[e] = (_Float16)acc[e];
      *(v8h*)(memb + base) = out;
    }
    __syncthreads();

    // ---- Phase 1: h1 = tanh(x_t@w_x^T + b_m + mem@u_h^T)  [128x256], 128 tiles
    const float* xin = inputs + (size_t)t * BATCH * HID;
    const int slot_new = t % KMEM;
    for (int i = 0; i < 4; ++i) {
      int tile = wave * 4 + i;
      int row0 = (tile >> 4) << 4;
      int n0   = (tile & 15) << 4;
      v8f c = {};
      for (int kb = 0; kb < HID; kb += 32) {
        v16h a  = load_a_f32(xin, HID, row0, kb, lane);
        v16h bf = load_b_packed(wxB, n0, kb, lane);
        c = wmma_f16(a, bf, c);
      }
      for (int kb = 0; kb < HID; kb += 32) {
        v16h a  = load_a_f16(memb, HID, row0, kb, lane);  // LDS
        v16h bf = load_b_packed(uB, n0, kb, lane);
        c = wmma_f16(a, bf, c);
      }
      const int half = lane >> 4, n = lane & 15;
      const int h = n0 + n;
      const float bmv = b_m[h];
#pragma unroll
      for (int j = 0; j < 8; ++j) {
        int m = row0 + j + 8 * half;
        float v = tanhf(c[j] + bmv);
        h1[m * HID + h] = (_Float16)v;                               // LDS
        ring[slot_new * (BATCH * HID) + m * HID + h] = (_Float16)v;  // global
      }
    }
    __syncthreads();

    // ---- Phase 2: gates = h1@w_ih^T + hx@w_hh^T + biases [128x1024], 512 tiles
    for (int i = 0; i < 16; ++i) {
      int tile = wave * 16 + i;
      int row0 = (tile >> 6) << 4;
      int n0   = (tile & 63) << 4;
      // prefetch next tile's first B panel into cache
      if (i + 1 < 16) {
        int nn0 = (((wave * 16 + i + 1) & 63) << 4);
        __builtin_prefetch(ihB + ((size_t)((nn0 >> 4) * (HID / 32)) << 9), 0, 1);
      }
      v8f c = {};
      for (int kb = 0; kb < HID; kb += 32) {
        v16h a  = load_a_f16(h1, HID, row0, kb, lane);   // LDS
        v16h bf = load_b_packed(ihB, n0, kb, lane);
        c = wmma_f16(a, bf, c);
      }
      for (int kb = 0; kb < HID; kb += 32) {
        v16h a  = load_a_f16(hx, HID, row0, kb, lane);   // LDS
        v16h bf = load_b_packed(hhB, n0, kb, lane);
        c = wmma_f16(a, bf, c);
      }
      const int half = lane >> 4, n = lane & 15;
      const int ng = n0 + n;
      const float bias = b_ih[ng] + b_hh[ng];
#pragma unroll
      for (int j = 0; j < 8; ++j)
        gates[(row0 + j + 8 * half) * G4H + ng] = c[j] + bias;
    }
    __syncthreads();

    // ---- Phase 3: LSTM elementwise (gate order i,f,g,o) + extra tanh.
    // 8-wide vectorized over consecutive h.
    for (int c0 = 0; c0 < (BATCH * HID) / (1024 * 8); ++c0) {
      int base = (tid + c0 * 1024) * 8;
      int b = base >> 8;
      int h0 = base & (HID - 1);
      const float* g = gates + b * G4H;
      v8f iv = *(const v8f*)(g + h0);
      v8f fv = *(const v8f*)(g + HID + h0);
      v8f gv = *(const v8f*)(g + 2 * HID + h0);
      v8f ov = *(const v8f*)(g + 3 * HID + h0);
      v8f cold = *(const v8f*)(cx + base);
      v8f cnew, hnew;
      v8h hxv;
#pragma unroll
      for (int e = 0; e < 8; ++e) {
        float cn = sigm(fv[e]) * cold[e] + sigm(iv[e]) * tanhf(gv[e]);
        float hn = tanhf(sigm(ov[e]) * tanhf(cn));
        cnew[e] = cn;
        hnew[e] = hn;
        hxv[e] = (_Float16)hn;
      }
      *(v8f*)(cx + base) = cnew;
      *(v8h*)(hx + base) = hxv;                    // LDS (read next step phase 2)
      if (length[b] == t) *(v8f*)(sel + base) = hnew;
    }
    __syncthreads();
  }
}

// ------------------------------------------------------------------
// Kernel 3: logits + log_softmax  (tiny: [128x256]x[256x5])
// ------------------------------------------------------------------
__global__ void mrnn_final(const char* __restrict__ ws, const float* __restrict__ w_out,
                           const float* __restrict__ b_out, float* __restrict__ out) {
  const int b = threadIdx.x;
  if (b >= BATCH) return;
  const float* sel = (const float*)(ws + WS_SEL) + b * HID;
  float logit[NOUT];
#pragma unroll
  for (int o = 0; o < NOUT; ++o) {
    float s = b_out[o];
    for (int h = 0; h < HID; ++h) s += sel[h] * w_out[o * HID + h];
    logit[o] = s;
  }
  float m = logit[0];
#pragma unroll
  for (int o = 1; o < NOUT; ++o) m = fmaxf(m, logit[o]);
  float z = 0.0f;
#pragma unroll
  for (int o = 0; o < NOUT; ++o) z += __expf(logit[o] - m);
  float lse = __logf(z);
#pragma unroll
  for (int o = 0; o < NOUT; ++o) out[b * NOUT + o] = logit[o] - m - lse;
}

extern "C" void kernel_launch(void* const* d_in, const int* in_sizes, int n_in,
                              void* d_out, int out_size, void* d_ws, size_t ws_size,
                              hipStream_t stream) {
  const float* inputs = (const float*)d_in[0];
  const int*   length = (const int*)d_in[1];
  const float* bd1    = (const float*)d_in[2];
  const float* w_x    = (const float*)d_in[3];
  const float* u_h    = (const float*)d_in[4];
  const float* b_m    = (const float*)d_in[5];
  const float* w_ih   = (const float*)d_in[6];
  const float* w_hh   = (const float*)d_in[7];
  const float* b_ih   = (const float*)d_in[8];
  const float* b_hh   = (const float*)d_in[9];
  const float* w_out  = (const float*)d_in[10];
  const float* b_out  = (const float*)d_in[11];
  float* out = (float*)d_out;
  char*  ws  = (char*)d_ws;

  mrnn_prep<<<512, 256, 0, stream>>>(bd1, w_x, u_h, w_ih, w_hh, ws);
  mrnn_scan<<<1, 1024, SM_TOTAL, stream>>>(inputs, length, b_m, b_ih, b_hh, ws);
  mrnn_final<<<1, 128, 0, stream>>>(ws, w_out, b_out, out);
}
